// MonarchLinear_57372173140182
// MI455X (gfx1250) — matrix operand order repaired and checked by reading, not compiled, not run on hardware
//
#include <hip/hip_runtime.h>

// ---------------------------------------------------------------------------
// MonarchLinear for MI455X (gfx1250, wave32, WMMA).
//   out = ((x @ W^T + b) blockdiag-mixed by inner) combined by outer
// Roofline: ~142 GFLOP, ~650 MB min traffic (512 MB output) -> ~28 us floor
// at 23.3 TB/s.  All three matmuls run v_wmma_f32_16x16x32_bf16 (f32 accum,
// bf16 operands converted on the fly).  K1 is LDS double-buffered (1 barrier
// per K-step, global loads overlap WMMA); K3 caches a 256x256 bf16 slab of
// `outer` in LDS and reuses it across 8 tokens (4 o-tiles total).
// f32->bf16 packing: v_add + v_perm_b32 (3 VALU per pair, round-half-up);
// K3's t-transpose packs u16 pairs with v_perm_b32 and stores b32 (DS ops
// halved vs scalar b16 stores).
// ---------------------------------------------------------------------------

typedef __attribute__((ext_vector_type(16))) __bf16 v16bf;
typedef __attribute__((ext_vector_type(8)))  float  v8f;
typedef unsigned short u16;
typedef unsigned int   u32;

#define TOKENS 2048
#define DIN    4096
#define NLAT   65536
#define GD     64      // GROUP_DIM
#define GM     4       // GROUP_MUL
#define IG     64      // IN_GROUPS
#define OG     1024    // OUT_GROUPS
#define MI     256     // GM*IG  (K of stage 3)

// ---- f32 -> bf16 (round-half-up), packed pair as raw u32 -------------------
// out[15:0] = bf16(a), out[31:16] = bf16(b): one v_perm_b32 picks the two
// rounded high halves.  sel 0x07060302: dst{b0,b1}=src1 bytes{2,3},
// dst{b2,b3}=src0 bytes{2,3}.
__device__ __forceinline__ u32 pk_bf16(float a, float b) {
  return __builtin_amdgcn_perm(__float_as_uint(b) + 0x8000u,
                               __float_as_uint(a) + 0x8000u,
                               0x07060302u);
}
__device__ __forceinline__ u16 f2bf(float a) {
  return (u16)((__float_as_uint(a) + 0x8000u) >> 16);
}

union FragU { uint4 q[2]; v16bf v; };

// A fragment 16x32 bf16 from LDS tile base[row][k] (row-major, ld elems).
// ISA 7.12.2: lanes 0-15 row=lane, K={0..7,16..23}; lanes 16-31 row=lane-16,
// K={8..15,24..31}.  Two ds_load_b128 per fragment.
__device__ __forceinline__ v16bf frag_a(const u16* base, int row0, int ld, int k0) {
  int lane = threadIdx.x & 31;
  int r  = lane & 15;
  int kb = (lane >> 4) << 3;               // 0 or 8
  const u16* p = base + (row0 + r) * ld + k0 + kb;
  FragU f;
  f.q[0] = *(const uint4*)(p);             // K kb..kb+7
  f.q[1] = *(const uint4*)(p + 16);        // K kb+16..kb+23
  return f.v;
}

// B fragment 32x16 bf16 from LDS stored column-major base[n][k] (ld elems).
// lanes 0-15 col=lane K={0..7,16..23}; lanes 16-31 col=lane-16 K={8..15,24..31}.
__device__ __forceinline__ v16bf frag_b(const u16* base, int col0, int ld, int k0) {
  int lane = threadIdx.x & 31;
  int n  = lane & 15;
  int kb = (lane >> 4) << 3;               // 0 or 8
  const u16* p = base + (col0 + n) * ld + k0 + kb;
  FragU f;
  f.q[0] = *(const uint4*)(p);             // K kb..kb+7
  f.q[1] = *(const uint4*)(p + 16);        // K kb+16..kb+23
  return f.v;
}

#define WMMA_BF16(A, B, C) \
  __builtin_amdgcn_wmma_f32_16x16x32_bf16(false, (A), false, (B), (short)0, (C), false, false)

// ===========================================================================
// Kernel 1: h[2048x4096] (bf16) = x @ W^T + b.   Both x and W are K-major.
// 128x128 block tile, K-steps of 32, LDS double-buffered: the next tile's
// global loads are issued before the current tile's WMMAs, one barrier/step.
// 8 waves: each wave 64(M) x 32(N) = 8 C-tiles (64 acc VGPRs).
// ===========================================================================
__global__ __launch_bounds__(256)
void monarch_k1_pre(const float* __restrict__ x, const float* __restrict__ w,
                    const float* __restrict__ bias, u16* __restrict__ h) {
  __shared__ __align__(16) u16 As[2][128 * 32];
  __shared__ __align__(16) u16 Bs[2][128 * 32];
  const int n0  = blockIdx.x * 128;
  const int m0  = blockIdx.y * 128;
  const int tid = threadIdx.x;
  const int wv  = tid >> 5;
  const int moff = (wv >> 2) * 64;         // 0,64
  const int noff = (wv & 3) * 32;          // 0,32,64,96

  v8f acc[4][2];
#pragma unroll
  for (int a = 0; a < 4; ++a)
#pragma unroll
    for (int c = 0; c < 2; ++c) acc[a][c] = (v8f){0,0,0,0,0,0,0,0};

  const int row = tid >> 1;                // 0..127
  const int c0  = (tid & 1) << 4;          // 0 or 16
  const float* pa_base = x + (size_t)(m0 + row) * DIN + c0;
  const float* pb_base = w + (size_t)(n0 + row) * DIN + c0;

  float4 ra[4], rb[4];
  auto load_regs = [&](int kk) {
    const float4* pa = (const float4*)(pa_base + kk);
    const float4* pb = (const float4*)(pb_base + kk);
#pragma unroll
    for (int q = 0; q < 4; ++q) { ra[q] = pa[q]; rb[q] = pb[q]; }
  };
  auto store_lds = [&](int buf) {
    *(uint4*)(As[buf] + row * 32 + c0)     = make_uint4(pk_bf16(ra[0].x,ra[0].y), pk_bf16(ra[0].z,ra[0].w),
                                                        pk_bf16(ra[1].x,ra[1].y), pk_bf16(ra[1].z,ra[1].w));
    *(uint4*)(As[buf] + row * 32 + c0 + 8) = make_uint4(pk_bf16(ra[2].x,ra[2].y), pk_bf16(ra[2].z,ra[2].w),
                                                        pk_bf16(ra[3].x,ra[3].y), pk_bf16(ra[3].z,ra[3].w));
    *(uint4*)(Bs[buf] + row * 32 + c0)     = make_uint4(pk_bf16(rb[0].x,rb[0].y), pk_bf16(rb[0].z,rb[0].w),
                                                        pk_bf16(rb[1].x,rb[1].y), pk_bf16(rb[1].z,rb[1].w));
    *(uint4*)(Bs[buf] + row * 32 + c0 + 8) = make_uint4(pk_bf16(rb[2].x,rb[2].y), pk_bf16(rb[2].z,rb[2].w),
                                                        pk_bf16(rb[3].x,rb[3].y), pk_bf16(rb[3].z,rb[3].w));
  };

  load_regs(0);
  store_lds(0);
  __syncthreads();

  int cur = 0;
  for (int kk = 0; kk < DIN; kk += 32) {
    const bool more = (kk + 32) < DIN;
    if (more) load_regs(kk + 32);          // global loads in flight over WMMAs

    v16bf af[4], bv[2];
#pragma unroll
    for (int mt = 0; mt < 4; ++mt) af[mt] = frag_a(As[cur], moff + mt * 16, 32, 0);
#pragma unroll
    for (int nt = 0; nt < 2; ++nt) bv[nt] = frag_b(Bs[cur], noff + nt * 16, 32, 0);
#pragma unroll
    for (int mt = 0; mt < 4; ++mt)
#pragma unroll
      for (int nt = 0; nt < 2; ++nt)
        acc[mt][nt] = WMMA_BF16(af[mt], bv[nt], acc[mt][nt]);

    if (more) {
      store_lds(cur ^ 1);                  // write the idle buffer
      __syncthreads();                     // one barrier per K-step
      cur ^= 1;
    }
  }

  // Epilogue: bias add, convert to bf16, store h.
  const int lane = tid & 31;
  const int nn = lane & 15;
  const int mh = (lane >> 4) << 3;         // 0 or 8
#pragma unroll
  for (int mt = 0; mt < 4; ++mt)
#pragma unroll
    for (int nt = 0; nt < 2; ++nt) {
      const int gj = n0 + noff + nt * 16 + nn;
      const float bvv = bias[gj];
#pragma unroll
      for (int r = 0; r < 8; ++r) {
        const int gb = m0 + moff + mt * 16 + mh + r;
        h[(size_t)gb * DIN + gj] = f2bf(acc[mt][nt][r] + bvv);
      }
    }
}

// ===========================================================================
// Kernel 2: t[b][mi][y] (bf16) = sum_x h[b][i*64+x] * inner[m][i][x][y].
// WG = (64 tokens, one i, all 4 m).  inner transposed into LDS [m][y][x].
// 8 waves: wave w -> m = w>>1, y-half = (w&1)*32; 64(M) x 32(N) per wave.
// ===========================================================================
__global__ __launch_bounds__(256)
void monarch_k2_blockdiag(const u16* __restrict__ h, const float* __restrict__ inner,
                          u16* __restrict__ tb) {
  __shared__ __align__(16) u16 Hs[64 * 64];        // [b][x]     8 KB
  __shared__ __align__(16) u16 Is[GM * 64 * 64];   // [m][y][x] 32 KB
  const int b0  = blockIdx.x * 64;
  const int i   = blockIdx.y;
  const int tid = threadIdx.x;

  { // h tile (already bf16): 64x64, coalesced b128 loads
    const int row = tid >> 2, cc = (tid & 3) << 4;
    const uint4* src = (const uint4*)(h + (size_t)(b0 + row) * DIN + i * GD + cc);
    uint4* dst = (uint4*)(Hs + row * 64 + cc);
    dst[0] = src[0];
    dst[1] = src[1];
  }
  { // inner[m][i] transposed: read rows of y coalesced, scatter to [y][x]
    const int xr = tid >> 2, y0 = (tid & 3) << 4;
#pragma unroll
    for (int m = 0; m < GM; ++m) {
      const float* src = inner + ((size_t)((m * IG + i) * GD + xr)) * GD + y0;
#pragma unroll
      for (int j = 0; j < 16; ++j)
        Is[m * 4096 + (y0 + j) * 64 + xr] = f2bf(src[j]);
    }
  }
  __syncthreads();

  const int wv = tid >> 5;
  const int m  = wv >> 1;
  const int yh = (wv & 1) * 32;

  v8f acc[4][2];
#pragma unroll
  for (int a = 0; a < 4; ++a)
#pragma unroll
    for (int c = 0; c < 2; ++c) acc[a][c] = (v8f){0,0,0,0,0,0,0,0};

#pragma unroll
  for (int ks = 0; ks < GD; ks += 32) {
    v16bf af[4], bv[2];
#pragma unroll
    for (int mt = 0; mt < 4; ++mt) af[mt] = frag_a(Hs, mt * 16, 64, ks);
#pragma unroll
    for (int nt = 0; nt < 2; ++nt) bv[nt] = frag_b(Is + m * 4096, yh + nt * 16, 64, ks);
#pragma unroll
    for (int mt = 0; mt < 4; ++mt)
#pragma unroll
      for (int nt = 0; nt < 2; ++nt)
        acc[mt][nt] = WMMA_BF16(af[mt], bv[nt], acc[mt][nt]);
  }

  const int lane = tid & 31;
  const int nn = lane & 15;
  const int mh = (lane >> 4) << 3;
  const int mi = m * IG + i;
#pragma unroll
  for (int mt = 0; mt < 4; ++mt)
#pragma unroll
    for (int nt = 0; nt < 2; ++nt) {
      const int y = yh + nt * 16 + nn;
#pragma unroll
      for (int r = 0; r < 8; ++r) {
        const int gb = b0 + mt * 16 + mh + r;
        tb[((size_t)gb * MI + mi) * GD + y] = f2bf(acc[mt][nt][r]);
      }
    }
}

// ===========================================================================
// Kernel 3: per token b:  out_b(o x y) = outer(o x 256) @ t_b(256 x 64).
// outer slab (256x256 bf16, 128 KB) cached in LDS once, reused for 8 tokens;
// only 4 o-tiles total.  t_b transposed into LDS [y][k] per token as the B
// operand (u16-pair packing via v_perm_b32, b32 stores).  LDS 160 KB ->
// 2 WGs / WGP.  8 waves: o-off = (w>>1)*64, y-half = (w&1)*32; 64x32 each.
// ===========================================================================
__global__ __launch_bounds__(256)
void monarch_k3_outer(const u16* __restrict__ tb, const float* __restrict__ outer,
                      float* __restrict__ out) {
  __shared__ __align__(16) u16 Os[256 * MI];  // [o][k] 128 KB
  __shared__ __align__(16) u16 Ts[GD  * MI];  // [y][k]  32 KB
  const int o0  = blockIdx.x * 256;
  const int tg  = blockIdx.y;                 // 8 tokens per WG
  const int tid = threadIdx.x;

  { // outer -> bf16 LDS, once; each thread converts one full 256-wide row
    const float* src = outer + (size_t)(o0 + tid) * MI;
    u16* dst = Os + tid * MI;
#pragma unroll
    for (int j = 0; j < 32; ++j) {
      float4 f0 = ((const float4*)src)[2 * j];
      float4 f1 = ((const float4*)src)[2 * j + 1];
      *(uint4*)(dst + j * 8) = make_uint4(pk_bf16(f0.x,f0.y), pk_bf16(f0.z,f0.w),
                                          pk_bf16(f1.x,f1.y), pk_bf16(f1.z,f1.w));
    }
  }

  const int wv = tid >> 5;
  const int ooff = (wv >> 1) * 64;            // 0,64,128,192
  const int yh   = (wv & 1) * 32;             // 0,32
  const int lane = tid & 31;
  const int nn = lane & 15;
  const int mh = (lane >> 4) << 3;

  for (int j = 0; j < 8; ++j) {
    const int b = tg * 8 + j;
    __syncthreads();                          // Os ready / prev compute done
    { // t_b [k][y] -> Ts [y][k]: thread owns a k-pair x 32 y values.
      // Pack (t[k+1][y], t[k][y]) with v_perm_b32, store b32: conflict-free,
      // half the DS ops of scalar b16 stores.  Same u16 layout in Ts.
      const int kp  = tid & 127;              // k-pair index: k = 2kp, 2kp+1
      const int yb  = (tid >> 7) * 32;        // y half: 0 or 32
      const u16* base = tb + (size_t)b * (MI * GD) + yb;
      const uint4* r0 = (const uint4*)(base + (size_t)(2 * kp)     * GD);
      const uint4* r1 = (const uint4*)(base + (size_t)(2 * kp + 1) * GD);
      u32* ts32 = (u32*)Ts;
#pragma unroll
      for (int q = 0; q < 4; ++q) {
        uint4 a4 = r0[q], b4 = r1[q];
        const u32 wa[4] = {a4.x, a4.y, a4.z, a4.w};
        const u32 wb[4] = {b4.x, b4.y, b4.z, b4.w};
#pragma unroll
        for (int e = 0; e < 8; ++e) {
          const int y = yb + q * 8 + e;
          // dst[15:0] = wa u16[e], dst[31:16] = wb u16[e]
          u32 v = __builtin_amdgcn_perm(wb[e >> 1], wa[e >> 1],
                                        (e & 1) ? 0x07060302u : 0x05040100u);
          ts32[y * (MI / 2) + kp] = v;
        }
      }
    }
    __syncthreads();

    v8f acc[4][2];
#pragma unroll
    for (int a = 0; a < 4; ++a)
#pragma unroll
      for (int c = 0; c < 2; ++c) acc[a][c] = (v8f){0,0,0,0,0,0,0,0};

#pragma unroll
    for (int ks = 0; ks < MI; ks += 32) {
      v16bf af[4], bv[2];
#pragma unroll
      for (int mt = 0; mt < 4; ++mt) af[mt] = frag_a(Os, ooff + mt * 16, MI, ks);
#pragma unroll
      for (int nt = 0; nt < 2; ++nt) bv[nt] = frag_b(Ts, yh + nt * 16, MI, ks);
#pragma unroll
      for (int mt = 0; mt < 4; ++mt)
#pragma unroll
        for (int nt = 0; nt < 2; ++nt)
          acc[mt][nt] = WMMA_BF16(af[mt], bv[nt], acc[mt][nt]);
    }

#pragma unroll
    for (int mt = 0; mt < 4; ++mt)
#pragma unroll
      for (int nt = 0; nt < 2; ++nt) {
        const int y = yh + nt * 16 + nn;
#pragma unroll
        for (int r = 0; r < 8; ++r) {
          const int o = o0 + ooff + mt * 16 + mh + r;
          out[(size_t)b * NLAT + (size_t)o * GD + y] = acc[mt][nt][r];
        }
      }
  }
}

// ===========================================================================
extern "C" void kernel_launch(void* const* d_in, const int* in_sizes, int n_in,
                              void* d_out, int out_size, void* d_ws, size_t ws_size,
                              hipStream_t stream) {
  const float* x     = (const float*)d_in[0];
  const float* pre_w = (const float*)d_in[1];
  const float* pre_b = (const float*)d_in[2];
  const float* inner = (const float*)d_in[3];
  const float* outer = (const float*)d_in[4];
  float* out = (float*)d_out;

  // workspace: h (bf16, 16 MB) then t (bf16, 64 MB)
  u16* h  = (u16*)d_ws;
  u16* tb = (u16*)((char*)d_ws + (size_t)TOKENS * DIN * sizeof(u16));

  monarch_k1_pre      <<<dim3(DIN / 128, TOKENS / 128), 256, 0, stream>>>(x, pre_w, pre_b, h);
  monarch_k2_blockdiag<<<dim3(TOKENS / 64, IG),          256, 0, stream>>>(h, inner, tb);
  monarch_k3_outer    <<<dim3(OG / 256, TOKENS / 8),     256, 0, stream>>>(tb, outer, out);
}
